// Block_20289425506613
// MI455X (gfx1250) — matrix operand order repaired and checked by reading, not compiled, not run on hardware
//
#include <hip/hip_runtime.h>
#include <math.h>

// ---------------------------------------------------------------------------
// MLA transformer block for MI455X (gfx1250, wave32, WMMA bf16).
// All GEMMs + both attention matmuls use v_wmma_f32_16x16x32_bf16.
// Operands pre-staged in global memory as bf16; LDS fills are pure DMA via
// GLOBAL_LOAD_ASYNC_TO_LDS_B128, double-buffered against the WMMA compute
// using partial waits (s_wait_asynccnt <= 4).
// ---------------------------------------------------------------------------

typedef __attribute__((ext_vector_type(16))) __bf16 bf16x16;
typedef __attribute__((ext_vector_type(8)))  float  floatx8;
typedef __attribute__((ext_vector_type(4)))  int    i32x4;

struct Frag { union { bf16x16 v; __bf16 h[16]; }; };
union U16x8 { uint4 q; __bf16 h[8]; };

// ---- async global->LDS copy (16 bytes per lane), with portable fallback ----
#if __has_builtin(__builtin_amdgcn_global_load_async_to_lds_b128)
#define HAS_ASYNC_LDS 1
__device__ __forceinline__ void copy16_g2l(const void* g, void* l) {
    __builtin_amdgcn_global_load_async_to_lds_b128(
        (__attribute__((address_space(1))) i32x4*)g,
        (__attribute__((address_space(3))) i32x4*)l, 0, 0);
}
#else
#define HAS_ASYNC_LDS 0
__device__ __forceinline__ void copy16_g2l(const void* g, void* l) {
    *(uint4*)l = *(const uint4*)g;
}
#endif

#if HAS_ASYNC_LDS
#if __has_builtin(__builtin_amdgcn_s_wait_asynccnt)
__device__ __forceinline__ void async_wait0() { __builtin_amdgcn_s_wait_asynccnt(0); }
__device__ __forceinline__ void async_wait4() { __builtin_amdgcn_s_wait_asynccnt(4); }
#else
__device__ __forceinline__ void async_wait0() { asm volatile("s_wait_asynccnt 0" ::: "memory"); }
__device__ __forceinline__ void async_wait4() { asm volatile("s_wait_asynccnt 4" ::: "memory"); }
#endif
#else
__device__ __forceinline__ void async_wait0() {}
__device__ __forceinline__ void async_wait4() {}
#endif

__device__ __forceinline__ floatx8 wmma_bf16(const Frag& a, const Frag& b, floatx8 c) {
    return __builtin_amdgcn_wmma_f32_16x16x32_bf16(false, a.v, false, b.v,
                                                   (short)0, c, false, false);
}

// A-fragment (16x32 bf16). p = &lds[row*stride + kb8], kb8 = (lane>>4)*8.
__device__ __forceinline__ void load_afrag(Frag& f, const __bf16* p) {
#pragma unroll
    for (int j = 0; j < 8; ++j) {
        int kk = (j < 4) ? (2 * j) : (16 + 2 * (j - 4));
        f.h[2 * j]     = p[kk];
        f.h[2 * j + 1] = p[kk + 1];
    }
}
// B-fragment (32x16 bf16): p = &lds[kb8*stride + n], u[s] = B[kk(s)][n] (table k order)
__device__ __forceinline__ void load_bfrag(Frag& f, const __bf16* p, int stride) {
#pragma unroll
    for (int j = 0; j < 16; ++j) {
        int kk = (j < 8) ? j : (16 + (j - 8));
        f.h[j] = p[kk * stride];
    }
}

// ---------------------------------------------------------------------------
// fp32 -> bf16 conversion (weights staging), n multiple of 4
// ---------------------------------------------------------------------------
__global__ __launch_bounds__(256) void cvt_f32_bf16(
    const float* __restrict__ in, __bf16* __restrict__ out, int n)
{
    int i = (blockIdx.x * 256 + threadIdx.x) * 4;
    if (i + 3 < n) {
        float4 f = *(const float4*)(in + i);
        out[i + 0] = (__bf16)f.x; out[i + 1] = (__bf16)f.y;
        out[i + 2] = (__bf16)f.z; out[i + 3] = (__bf16)f.w;
    }
}

// W_kr (1024x64) -> zero-padded 1024x128 bf16 panel, so the GEMM never needs
// an N-edge guard in its streaming loop.
__global__ __launch_bounds__(256) void cvt_pad_kr(
    const float* __restrict__ in, __bf16* __restrict__ out)
{
    int i = blockIdx.x * 256 + threadIdx.x;      // over 1024*128
    int k = i >> 7, n = i & 127;
    out[i] = (n < 64) ? (__bf16)in[k * 64 + n] : (__bf16)0.f;
}

// ---------------------------------------------------------------------------
// Generic GEMM:  C[M,N] = A[M,K](bf16) * B[K,N](bf16) + bias
// flag bits: 1 = gelu(tanh), 2 = fp32 residual add (stride ldc), 4 = bf16 out
// oscale applied after bias/gelu (folds 1/sqrt(DK) into q).
// block tile 128x128, k-step 32, 8 waves (2x4), wave tile 64x32.
// Double-buffered LDS: tile k+1 streams in via async DMA while tile k runs
// on the WMMA pipe; partial wait (asynccnt<=4) releases only tile k's batch.
// B must be readable for the full 128-wide panel (kR weight is pre-padded).
// ---------------------------------------------------------------------------
#define ASTR 40
#define BSTR 136

__global__ __launch_bounds__(256) void gemm_bf16_wmma(
    const __bf16* __restrict__ A, int lda,
    const __bf16* __restrict__ B, int ldb,
    const float* __restrict__ bias,
    const float* __restrict__ res,
    float* __restrict__ C, __bf16* __restrict__ Cbf, int ldc,
    int M, int N, int K, int flag, float oscale)
{
    __shared__ __bf16 sA[2][128 * ASTR];   // [m][k]
    __shared__ __bf16 sB[2][32 * BSTR];    // [k][n]

    int tid   = threadIdx.x;
    int lane  = tid & 31, wave = tid >> 5;
    int l16   = lane & 15, lhalf = lane >> 4;
    int kb8   = lhalf * 8;
    int wm    = wave & 1, wn = wave >> 1;       // 2 x 4 wave grid
    int bm    = blockIdx.y * 128, bn = blockIdx.x * 128;

    const floatx8 z8 = {0.f,0.f,0.f,0.f,0.f,0.f,0.f,0.f};
    floatx8 acc[4][2];
#pragma unroll
    for (int i = 0; i < 4; ++i)
#pragma unroll
        for (int j = 0; j < 2; ++j) acc[i][j] = z8;

    // per-thread staging coordinates (4 x 16B async ops per tile)
    int ar0 = (tid * 8) >> 5,           ac0 = (tid * 8) & 31;
    int ar1 = ((256 + tid) * 8) >> 5,   ac1 = ((256 + tid) * 8) & 31;
    int br0 = (tid * 8) >> 7,           bc0 = (tid * 8) & 127;
    int br1 = ((256 + tid) * 8) >> 7,   bc1 = ((256 + tid) * 8) & 127;

    auto stage = [&](int buf, int k0) {
        copy16_g2l(A + (size_t)(bm + ar0) * lda + k0 + ac0, &sA[buf][ar0 * ASTR + ac0]);
        copy16_g2l(A + (size_t)(bm + ar1) * lda + k0 + ac1, &sA[buf][ar1 * ASTR + ac1]);
        copy16_g2l(B + (size_t)(k0 + br0) * ldb + bn + bc0, &sB[buf][br0 * BSTR + bc0]);
        copy16_g2l(B + (size_t)(k0 + br1) * ldb + bn + bc1, &sB[buf][br1 * BSTR + bc1]);
    };

    int nk = K >> 5;
    stage(0, 0);                                // prologue: tile 0 -> buffer 0
    for (int kt = 0; kt < nk; ++kt) {
        int cb = kt & 1;
        if (kt + 1 < nk) {
            stage(cb ^ 1, (kt + 1) << 5);       // stream next tile
            async_wait4();                       // only tile kt's batch must land
        } else {
            async_wait0();
        }
        __syncthreads();

        Frag af[4], bfr[2];
#pragma unroll
        for (int mi = 0; mi < 4; ++mi)
            load_afrag(af[mi], &sA[cb][(wm * 64 + mi * 16 + l16) * ASTR + kb8]);
#pragma unroll
        for (int ni = 0; ni < 2; ++ni)
            load_bfrag(bfr[ni], &sB[cb][kb8 * BSTR + wn * 32 + ni * 16 + l16], BSTR);
#pragma unroll
        for (int mi = 0; mi < 4; ++mi)
#pragma unroll
            for (int ni = 0; ni < 2; ++ni)
                acc[mi][ni] = wmma_bf16(af[mi], bfr[ni], acc[mi][ni]);

        __syncthreads();                         // done reading cb before reuse
    }

    // epilogue: C/D layout lane=N (l16), vgpr v -> M = v + 8*(lane>>4)
#pragma unroll
    for (int mi = 0; mi < 4; ++mi) {
#pragma unroll
        for (int ni = 0; ni < 2; ++ni) {
            int n = bn + wn * 32 + ni * 16 + l16;
            if (n >= N) continue;
            int mbase = bm + wm * 64 + mi * 16 + 8 * lhalf;
            float bv = bias ? bias[n] : 0.f;
#pragma unroll
            for (int v = 0; v < 8; ++v) {
                int m = mbase + v;
                if (m >= M) continue;
                float val = acc[mi][ni][v] + bv;
                if (flag & 1) {
                    float t = 0.7978845608028654f * (val + 0.044715f * val * val * val);
                    val = 0.5f * val * (1.f + tanhf(t));
                }
                val *= oscale;
                if (flag & 4) {
                    Cbf[(size_t)m * ldc + n] = (__bf16)val;
                } else {
                    if (flag & 2) val += res[(size_t)m * ldc + n];
                    C[(size_t)m * ldc + n] = val;
                }
            }
        }
    }
}

// ---------------------------------------------------------------------------
// RMSNorm: one block per row, bf16 output (consumed only by GEMMs)
// ---------------------------------------------------------------------------
__global__ __launch_bounds__(256) void rmsnorm_kernel(
    const float* __restrict__ x, const float* __restrict__ w,
    __bf16* __restrict__ o, int cols)
{
    int row = blockIdx.x;
    const float* xr = x + (size_t)row * cols;
    float s = 0.f;
    for (int c = threadIdx.x; c < cols; c += 256) { float v = xr[c]; s += v * v; }
    __shared__ float red[256];
    red[threadIdx.x] = s; __syncthreads();
    for (int st = 128; st > 0; st >>= 1) {
        if (threadIdx.x < st) red[threadIdx.x] += red[threadIdx.x + st];
        __syncthreads();
    }
    float r = rsqrtf(red[0] / (float)cols + 1e-6f);
    for (int c = threadIdx.x; c < cols; c += 256)
        o[(size_t)row * cols + c] = (__bf16)(xr[c] * r * w[c]);
}

// ---------------------------------------------------------------------------
// RoPE (reference-faithful: freq = t+1, pair-repeated theta over c), fp32 in,
// scaled bf16 out (scale folds attention 1/sqrt(DK) for the q branch).
// ---------------------------------------------------------------------------
__global__ void rope_kernel(const float* __restrict__ in, __bf16* __restrict__ out,
                            int rows, int c, int T, float scale)
{
    int half = c >> 1;
    long long total = (long long)rows * half;
    for (long long i = (long long)blockIdx.x * blockDim.x + threadIdx.x;
         i < total; i += (long long)gridDim.x * blockDim.x) {
        int row = (int)(i / half);
        int j   = (int)(i - (long long)row * half);
        int t   = row % T;
        float theta = __expf(-2.0f * (float)j / (float)c * 9.210340371976184f); // ln(1e4)
        float ang = (float)(t + 1) * theta;
        float sn, cs;
        __sincosf(ang, &sn, &cs);
        size_t o = (size_t)row * c + 2 * j;
        float x0 = in[o], x1 = in[o + 1];
        out[o]     = (__bf16)((x0 * cs - x1 * sn) * scale);
        out[o + 1] = (__bf16)((x1 * cs + x0 * sn) * scale);
    }
}

// ---------------------------------------------------------------------------
// Flash-style causal MLA attention, all-bf16 inputs (q pre-scaled), bf16 out.
// grid = (T/64, NH, B), block = 128 (4 waves); wave w owns 16 q rows.
// Per 32-key tile: S = Q*K^T (8 wmma), online softmax, P*V (4 wmma).
// ---------------------------------------------------------------------------
#define QSTR 136
#define KSTR 40
#define VSTR 72
#define PSTR 36

__global__ __launch_bounds__(128) void mla_attention(
    const __bf16* __restrict__ qbuf, const __bf16* __restrict__ qR,
    const __bf16* __restrict__ kv,   const __bf16* __restrict__ kR,
    __bf16* __restrict__ out)
{
    const int T = 1024;
    __shared__ __bf16 sQ [64 * QSTR];     // [qrow][d]
    __shared__ __bf16 sKt[128 * KSTR];    // [d][key]   (K^T)
    __shared__ __bf16 sV [32 * VSTR];     // [key][dim]
    __shared__ __bf16 sP [4 * 16 * PSTR]; // per-wave 16x32

    int qb = blockIdx.x, nh = blockIdx.y, b = blockIdx.z;
    int tid  = threadIdx.x;
    int lane = tid & 31, wave = tid >> 5;
    int l16  = lane & 15, lhalf = lane >> 4, kb8 = lhalf * 8;
    size_t rb = (size_t)b * T;

    // stage Q tile (64 x 128): pure 16B async copies (scaling folded in)
#pragma unroll
    for (int i = 0; i < 8; ++i) {
        int idx = (i * 128 + tid) * 8;
        int r = idx >> 7, d = idx & 127;
        int t = qb * 64 + r;
        const __bf16* src = (d < 64) ? qbuf + (rb + t) * 1024 + nh * 64 + d
                                     : qR   + (rb + t) * 1024 + nh * 64 + (d - 64);
        copy16_g2l(src, &sQ[r * QSTR + d]);
    }
    async_wait0();
    __syncthreads();

    Frag aQ[4];
#pragma unroll
    for (int dk = 0; dk < 4; ++dk)
        load_afrag(aQ[dk], &sQ[(wave * 16 + l16) * QSTR + dk * 32 + kb8]);

    const floatx8 z8 = {0.f,0.f,0.f,0.f,0.f,0.f,0.f,0.f};
    floatx8 O[4];
#pragma unroll
    for (int j = 0; j < 4; ++j) O[j] = z8;
    float mrow[8], lrow[8];
#pragma unroll
    for (int v = 0; v < 8; ++v) { mrow[v] = -INFINITY; lrow[v] = 0.f; }

    int ktend = qb * 2 + 2;                 // causal: skip tiles above diagonal
    for (int kt = 0; kt < ktend; ++kt) {
        __syncthreads();
        // K tile 32x128, stored transposed [d][key]; vector load + scalar scatter
#pragma unroll
        for (int i = 0; i < 4; ++i) {
            int cc = i * 128 + tid;               // 512 chunks of 8 halfs
            int key = cc >> 4, c8 = (cc & 15) * 8;
            int t = kt * 32 + key;
            U16x8 u;
            u.q = (c8 < 64)
                ? *(const uint4*)(kv + (rb + t) * 2048 + nh * 64 + c8)
                : *(const uint4*)(kR + (rb + t) * 64 + (c8 - 64));
#pragma unroll
            for (int j = 0; j < 8; ++j) sKt[(c8 + j) * KSTR + key] = u.h[j];
        }
        // V tile 32x64: pure async copies
#pragma unroll
        for (int i = 0; i < 2; ++i) {
            int idx = (i * 128 + tid) * 8;
            int r = idx >> 6, d = idx & 63;
            int t = kt * 32 + r;
            copy16_g2l(kv + (rb + t) * 2048 + 1024 + nh * 64 + d, &sV[r * VSTR + d]);
        }
        async_wait0();
        __syncthreads();

        // S = Q * K^T  (two 16-key column tiles)
        floatx8 s0 = z8, s1 = z8;
#pragma unroll
        for (int dk = 0; dk < 4; ++dk) {
            Frag b0, b1;
            load_bfrag(b0, &sKt[(dk * 32 + kb8) * KSTR + 0 + l16], KSTR);
            load_bfrag(b1, &sKt[(dk * 32 + kb8) * KSTR + 16 + l16], KSTR);
            s0 = wmma_bf16(aQ[dk], b0, s0);
            s1 = wmma_bf16(aQ[dk], b1, s1);
        }

        // online softmax; row m lives in vgpr v across one 16-lane half
        int qr0 = qb * 64 + wave * 16 + 8 * lhalf;
        __bf16* pbase = &sP[wave * 16 * PSTR];
#pragma unroll
        for (int v = 0; v < 8; ++v) {
            int qr = qr0 + v;
            float x0 = s0[v]; if (kt * 32 + l16 > qr)      x0 = -INFINITY;
            float x1 = s1[v]; if (kt * 32 + 16 + l16 > qr) x1 = -INFINITY;
            float t = fmaxf(x0, x1);
#pragma unroll
            for (int o = 8; o >= 1; o >>= 1) t = fmaxf(t, __shfl_xor(t, o, 32));
            float mn = fmaxf(mrow[v], t);
            float p0 = __expf(x0 - mn), p1 = __expf(x1 - mn);
            float rs = p0 + p1;
#pragma unroll
            for (int o = 8; o >= 1; o >>= 1) rs += __shfl_xor(rs, o, 32);
            float alpha = __expf(mrow[v] - mn);
            lrow[v] = lrow[v] * alpha + rs;
            mrow[v] = mn;
#pragma unroll
            for (int j = 0; j < 4; ++j) O[j][v] *= alpha;
            int prow = v + 8 * lhalf;
            pbase[prow * PSTR + l16]      = (__bf16)p0;
            pbase[prow * PSTR + 16 + l16] = (__bf16)p1;
        }
        __builtin_amdgcn_wave_barrier();      // keep LDS P write->read ordered

        // O += P * V  (k = 32 keys, 4 dim tiles)
        Frag aP;
        load_afrag(aP, &pbase[l16 * PSTR + kb8]);
#pragma unroll
        for (int j = 0; j < 4; ++j) {
            Frag bv;
            load_bfrag(bv, &sV[kb8 * VSTR + j * 16 + l16], VSTR);
            O[j] = wmma_bf16(aP, bv, O[j]);
        }
    }

    // normalize and write head-merged bf16 [b, t, nh*64 + d]
#pragma unroll
    for (int v = 0; v < 8; ++v) {
        float inv = 1.f / lrow[v];
        int t = qb * 64 + wave * 16 + v + 8 * lhalf;
#pragma unroll
        for (int j = 0; j < 4; ++j)
            out[(rb + t) * 1024 + nh * 64 + j * 16 + l16] = (__bf16)(O[j][v] * inv);
    }
}

// ---------------------------------------------------------------------------
// Host-side launch
// ---------------------------------------------------------------------------
extern "C" void kernel_launch(void* const* d_in, const int* in_sizes, int n_in,
                              void* d_out, int out_size, void* d_ws, size_t ws_size,
                              hipStream_t stream)
{
    (void)in_sizes; (void)n_in; (void)out_size; (void)ws_size;
    const float* x     = (const float*)d_in[0];
    const float* rms1  = (const float*)d_in[1];
    const float* rms2  = (const float*)d_in[2];
    const float* W_dkv = (const float*)d_in[3];
    const float* b_dkv = (const float*)d_in[4];
    const float* W_kr  = (const float*)d_in[5];
    const float* b_kr  = (const float*)d_in[6];
    const float* W_qr  = (const float*)d_in[7];
    const float* b_qr  = (const float*)d_in[8];
    const float* W_kv  = (const float*)d_in[9];
    const float* b_kv  = (const float*)d_in[10];
    const float* W_q   = (const float*)d_in[11];
    const float* b_q   = (const float*)d_in[12];
    const float* W_o   = (const float*)d_in[13];
    const float* b_o   = (const float*)d_in[14];
    const float* W_f1  = (const float*)d_in[15];
    const float* b_f1  = (const float*)d_in[16];
    const float* W_f2  = (const float*)d_in[17];
    const float* b_f2  = (const float*)d_in[18];
    float* out = (float*)d_out;

    const int M = 4096;                          // B*T
    char* cur = (char*)d_ws;
    auto alloc = [&](size_t bytes) {
        char* p = cur;
        cur += (bytes + 255) & ~(size_t)255;
        return (void*)p;
    };
    // fp32 scratch
    float* qRf  = (float*)alloc((size_t)M * 1024 * 4);
    float* kRf  = (float*)alloc((size_t)M * 64 * 4);
    float* hbuf = (float*)alloc((size_t)M * 1024 * 4);
    // bf16 activations
    __bf16* xnbf   = (__bf16*)alloc((size_t)M * 1024 * 2);   // also h2n
    __bf16* compbf = (__bf16*)alloc((size_t)M * 1024 * 2);
    __bf16* qbf    = (__bf16*)alloc((size_t)M * 1024 * 2);
    __bf16* qRbf   = (__bf16*)alloc((size_t)M * 1024 * 2);
    __bf16* kRbf   = (__bf16*)alloc((size_t)M * 64 * 2);
    __bf16* kvbf   = (__bf16*)alloc((size_t)M * 2048 * 2);
    __bf16* attnbf = (__bf16*)alloc((size_t)M * 1024 * 2);
    __bf16* ffhbf  = (__bf16*)alloc((size_t)M * 4096 * 2);
    // bf16 weights (wkr zero-padded 1024x64 -> 1024x128)
    __bf16* wdkv = (__bf16*)alloc(1048576 * 2);
    __bf16* wkr  = (__bf16*)alloc(131072 * 2);
    __bf16* wqr  = (__bf16*)alloc(1048576 * 2);
    __bf16* wkv  = (__bf16*)alloc(1048576 * 2);
    __bf16* wq   = (__bf16*)alloc(524288 * 2);
    __bf16* wo   = (__bf16*)alloc(1048576 * 2);
    __bf16* wf1  = (__bf16*)alloc(4194304 * 2);
    __bf16* wf2  = (__bf16*)alloc(4194304 * 2);
    __bf16* h2nbf = xnbf;

    auto cvt = [&](const float* src, __bf16* dst, int n) {
        cvt_f32_bf16<<<(n / 4 + 255) / 256, 256, 0, stream>>>(src, dst, n);
    };
    cvt(W_dkv, wdkv, 1048576);
    cvt_pad_kr<<<131072 / 256, 256, 0, stream>>>(W_kr, wkr);
    cvt(W_qr,  wqr,  1048576);
    cvt(W_kv,  wkv,  1048576);
    cvt(W_q,   wq,   524288);
    cvt(W_o,   wo,   1048576);
    cvt(W_f1,  wf1,  4194304);
    cvt(W_f2,  wf2,  4194304);

    // 1) xn = rmsnorm(x) -> bf16
    rmsnorm_kernel<<<M, 256, 0, stream>>>(x, rms1, xnbf, 1024);
    // 2) comp = xn @ W_dkv + b_dkv -> bf16
    gemm_bf16_wmma<<<dim3(8, 32), 256, 0, stream>>>(xnbf, 1024, wdkv, 1024, b_dkv,
        nullptr, nullptr, compbf, 1024, M, 1024, 1024, 4, 1.f);
    // 3) kR = comp @ W_kr + b_kr -> fp32 (rope input); B panel padded to 128
    gemm_bf16_wmma<<<dim3(1, 32), 256, 0, stream>>>(compbf, 1024, wkr, 128, b_kr,
        nullptr, kRf, nullptr, 64, M, 64, 1024, 0, 1.f);
    // 4) qR = comp @ W_qr + b_qr -> fp32 (rope input)
    gemm_bf16_wmma<<<dim3(8, 32), 256, 0, stream>>>(compbf, 1024, wqr, 1024, b_qr,
        nullptr, qRf, nullptr, 1024, M, 1024, 1024, 0, 1.f);
    // 5) RoPE -> bf16 (q branch folds 1/sqrt(DK))
    rope_kernel<<<(M * 32 + 255) / 256, 256, 0, stream>>>(kRf, kRbf, M, 64, 1024, 1.f);
    rope_kernel<<<(M * 512 + 255) / 256, 256, 0, stream>>>(qRf, qRbf, M, 1024, 1024, 0.125f);
    // 6) kv = cKV @ W_kv + b_kv -> bf16   (cKV = comp[:, :512])
    gemm_bf16_wmma<<<dim3(16, 32), 256, 0, stream>>>(compbf, 1024, wkv, 2048, b_kv,
        nullptr, nullptr, kvbf, 2048, M, 2048, 512, 4, 1.f);
    // 7) q = cq @ W_q + b_q -> bf16, pre-scaled by 1/sqrt(DK)  (cq = comp[:, 512:])
    gemm_bf16_wmma<<<dim3(8, 32), 256, 0, stream>>>(compbf + 512, 1024, wq, 1024, b_q,
        nullptr, nullptr, qbf, 1024, M, 1024, 512, 4, 0.125f);
    // 8) attention -> bf16
    mla_attention<<<dim3(16, 16, 4), 128, 0, stream>>>(qbf, qRbf, kvbf, kRbf, attnbf);
    // 9) h = x + attn @ W_o + b_o -> fp32
    gemm_bf16_wmma<<<dim3(8, 32), 256, 0, stream>>>(attnbf, 1024, wo, 1024, b_o,
        x, hbuf, nullptr, 1024, M, 1024, 1024, 2, 1.f);
    // 10) h2n = rmsnorm(h) -> bf16
    rmsnorm_kernel<<<M, 256, 0, stream>>>(hbuf, rms2, h2nbf, 1024);
    // 11) ffh = gelu(h2n @ W_f1 + b_f1) -> bf16
    gemm_bf16_wmma<<<dim3(32, 32), 256, 0, stream>>>(h2nbf, 1024, wf1, 4096, b_f1,
        nullptr, nullptr, ffhbf, 4096, M, 4096, 1024, 5, 1.f);
    // 12) out = h + ffh @ W_f2 + b_f2 -> fp32
    gemm_bf16_wmma<<<dim3(8, 32), 256, 0, stream>>>(ffhbf, 4096, wf2, 1024, b_f2,
        hbuf, out, nullptr, 1024, M, 1024, 4096, 2, 1.f);
}
